// GraphProp_36180804502163
// MI455X (gfx1250) — matrix operand order repaired and checked by reading, not compiled, not run on hardware
//
#include <hip/hip_runtime.h>

typedef __attribute__((ext_vector_type(16))) __bf16 v16bf;
typedef __attribute__((ext_vector_type(8)))  float  v8f;
typedef unsigned short u16;
typedef unsigned int   u32;

union Frag16 {
    uint4 q[2];
    v16bf v;
    u16   s[16];
};

__device__ __forceinline__ u16 f2bf(float f) {
    u32 u = __float_as_uint(f);
    u32 r = u + 0x7FFFu + ((u >> 16) & 1u);   // round-to-nearest-even
    return (u16)(r >> 16);
}

// Pack float4 -> 4 bf16 in a uint2 (for one ds_store_b64)
__device__ __forceinline__ uint2 f4_to_bf4(float4 v) {
    u32 lo = ((u32)f2bf(v.y) << 16) | (u32)f2bf(v.x);
    u32 hi = ((u32)f2bf(v.w) << 16) | (u32)f2bf(v.z);
    return make_uint2(lo, hi);
}

__device__ __forceinline__ float sigmoidf(float x) {
    return 1.0f / (1.0f + __expf(-x));
}

// ---------------------------------------------------------------------------
// Zero the mailbox a[N,128]
// ---------------------------------------------------------------------------
__global__ __launch_bounds__(256) void zero_kernel(float4* p, int n4) {
    int i = blockIdx.x * blockDim.x + threadIdx.x;
    if (i < n4) p[i] = make_float4(0.f, 0.f, 0.f, 0.f);
}

// ---------------------------------------------------------------------------
// Edge kernel: per 16-edge tile, feat=[hv[dst]|hv[src]|he] (16x160) times
// Wm (160x128) via bf16 WMMA; atomic scatter (msg+bm) into a[dst].
// 256 threads = 8 waves; wave w owns output columns [16w,16w+16).
// ---------------------------------------------------------------------------
__global__ __launch_bounds__(256) void edge_kernel(
    const float* __restrict__ hv, const float* __restrict__ he,
    const int* __restrict__ src, const int* __restrict__ dst,
    const float* __restrict__ Wm_t, const float* __restrict__ bm_t,
    float* __restrict__ a, int n_tiles)
{
    extern __shared__ char smem[];
    u16* sFeat = (u16*)smem;                     // [16][160] bf16
    int* sIdx  = (int*)(smem + 16 * 160 * 2);    // dst[16], src[16]

    const int tid     = threadIdx.x;
    const int lane    = tid & 31;
    const int wave    = tid >> 5;
    const int halfgrp = lane >> 4;   // 0: lanes 0-15, 1: lanes 16-31
    const int lrow    = lane & 15;
    const int ncol    = wave * 16 + lrow;        // output column (0..127)

    // Preload this wave's 5 B fragments of Wm (loop-invariant). B layout:
    // half h -> K = 32c + (h < 8 ? halfgrp*8 + h : 16 + halfgrp*8 + h - 8), N = ncol.
    Frag16 Bf[5];
#pragma unroll
    for (int c = 0; c < 5; ++c)
#pragma unroll
        for (int h = 0; h < 16; ++h) {
            int k = 32 * c + (h < 8 ? halfgrp * 8 + h : 16 + halfgrp * 8 + (h - 8));
            Bf[c].s[h] = f2bf(Wm_t[k * 128 + ncol]);
        }
    const float bmv = bm_t[ncol];

    for (int tile = blockIdx.x; tile < n_tiles; tile += gridDim.x) {
        const int ebase = tile * 16;
        __syncthreads();                         // protect sFeat/sIdx reuse
        if (tid < 16) {
            sIdx[tid]      = dst[ebase + tid];
            sIdx[16 + tid] = src[ebase + tid];
        }
        __syncthreads();

        // Stage feat tile (16x160 f32 -> bf16), vectorized: 640 float4 loads
        for (int i = tid; i < 16 * 40; i += 256) {
            int f = i / 40, q = i - f * 40;
            int k = q * 4;
            float4 v;
            if (k < 64)        v = *(const float4*)&hv[(size_t)sIdx[f] * 64 + k];
            else if (k < 128)  v = *(const float4*)&hv[(size_t)sIdx[16 + f] * 64 + (k - 64)];
            else               v = *(const float4*)&he[(size_t)(ebase + f) * 32 + (k - 128)];
            *(uint2*)(sFeat + f * 160 + k) = f4_to_bf4(v);
        }
        __syncthreads();

        v8f cacc = {};
#pragma unroll
        for (int c = 0; c < 5; ++c) {
            // A layout: halves 0-7 -> K = 32c + halfgrp*8 + 0..7,
            //           halves 8-15 -> K = 32c + 16 + halfgrp*8 + 0..7
            Frag16 Af;
            const u16* rp = sFeat + lrow * 160 + 32 * c + halfgrp * 8;
            Af.q[0] = *(const uint4*)rp;
            Af.q[1] = *(const uint4*)(rp + 16);
            cacc = __builtin_amdgcn_wmma_f32_16x16x32_bf16(
                false, Af.v, false, Bf[c].v, (short)0, cacc, false, false);
        }

        // Scatter: C layout VGPR r -> M = r + halfgrp*8, N = lrow
#pragma unroll
        for (int r = 0; r < 8; ++r) {
            int m = r + halfgrp * 8;
            int d = sIdx[m];
            atomicAdd(&a[(size_t)d * 128 + ncol], cacc[r] + bmv);
        }
    }
}

// ---------------------------------------------------------------------------
// GRU kernel: per 16-node tile, gi = a·Wih^T + bih (K=128, Nout=192),
// gh = h·Whh^T + bhh (K=64, Nout=192) via bf16 WMMA, then f32 gate math.
// In-place safe (each node's output depends only on its own staged row).
// ---------------------------------------------------------------------------
__global__ __launch_bounds__(256) void gru_kernel(
    const float* __restrict__ a, const float* __restrict__ hv_in,
    float* __restrict__ hv_out,
    const float* __restrict__ Wih_t, const float* __restrict__ Whh_t,
    const float* __restrict__ bih_t, const float* __restrict__ bhh_t,
    int n_tiles)
{
    extern __shared__ char smem[];
    u16*   sWih = (u16*)smem;               // [192][128] bf16
    u16*   sWhh = sWih + 192 * 128;         // [192][64]  bf16
    u16*   sA   = sWhh + 192 * 64;          // [16][128]  bf16
    u16*   sH   = sA + 16 * 128;            // [16][64]   bf16
    float* sGi  = (float*)(sH + 16 * 64);   // [16][192]  f32
    float* sGh  = sGi + 16 * 192;           // [16][192]  f32

    const int tid     = threadIdx.x;
    const int lane    = tid & 31;
    const int wave    = tid >> 5;
    const int halfgrp = lane >> 4;
    const int lrow    = lane & 15;

    // Stage weights once per block (f32 -> bf16), float4-vectorized
    for (int i = tid; i < (192 * 128) / 4; i += 256)
        *(uint2*)(sWih + i * 4) = f4_to_bf4(*(const float4*)&Wih_t[i * 4]);
    for (int i = tid; i < (192 * 64) / 4; i += 256)
        *(uint2*)(sWhh + i * 4) = f4_to_bf4(*(const float4*)&Whh_t[i * 4]);

    for (int tile = blockIdx.x; tile < n_tiles; tile += gridDim.x) {
        const int nbase = tile * 16;
        __syncthreads();
        for (int i = tid; i < (16 * 128) / 4; i += 256)
            *(uint2*)(sA + i * 4) = f4_to_bf4(*(const float4*)&a[(size_t)nbase * 128 + i * 4]);
        for (int i = tid; i < (16 * 64) / 4; i += 256)
            *(uint2*)(sH + i * 4) = f4_to_bf4(*(const float4*)&hv_in[(size_t)nbase * 64 + i * 4]);
        __syncthreads();

        // 12 output tiles of 16 cols over 8 waves
        for (int jt = wave; jt < 12; jt += 8) {
            const int j = jt * 16 + lrow;

            v8f cg = {};
#pragma unroll
            for (int c = 0; c < 4; ++c) {
                Frag16 Af, Bf;
                const u16* rp = sA + lrow * 128 + 32 * c + halfgrp * 8;
                Af.q[0] = *(const uint4*)rp;
                Af.q[1] = *(const uint4*)(rp + 16);
                const u16* bp = sWih + j * 128 + 32 * c + halfgrp * 8;
                Bf.q[0] = *(const uint4*)bp;
                Bf.q[1] = *(const uint4*)(bp + 16);
                cg = __builtin_amdgcn_wmma_f32_16x16x32_bf16(
                    false, Af.v, false, Bf.v, (short)0, cg, false, false);
            }
#pragma unroll
            for (int r = 0; r < 8; ++r) {
                int m = r + halfgrp * 8;
                sGi[m * 192 + j] = cg[r] + bih_t[j];
            }

            v8f ch = {};
#pragma unroll
            for (int c = 0; c < 2; ++c) {
                Frag16 Af, Bf;
                const u16* rp = sH + lrow * 64 + 32 * c + halfgrp * 8;
                Af.q[0] = *(const uint4*)rp;
                Af.q[1] = *(const uint4*)(rp + 16);
                const u16* bp = sWhh + j * 64 + 32 * c + halfgrp * 8;
                Bf.q[0] = *(const uint4*)bp;
                Bf.q[1] = *(const uint4*)(bp + 16);
                ch = __builtin_amdgcn_wmma_f32_16x16x32_bf16(
                    false, Af.v, false, Bf.v, (short)0, ch, false, false);
            }
#pragma unroll
            for (int r = 0; r < 8; ++r) {
                int m = r + halfgrp * 8;
                sGh[m * 192 + j] = ch[r] + bhh_t[j];
            }
        }
        __syncthreads();

        // Element-wise GRU gates (torch order r, z, n)
        for (int i = tid; i < 16 * 64; i += 256) {
            int f = i >> 6, q = i & 63;
            float ir  = sGi[f * 192 + q];
            float iz  = sGi[f * 192 + 64 + q];
            float inn = sGi[f * 192 + 128 + q];
            float hr  = sGh[f * 192 + q];
            float hz  = sGh[f * 192 + 64 + q];
            float hn  = sGh[f * 192 + 128 + q];
            float h   = hv_in[(size_t)(nbase + f) * 64 + q];
            float rr  = sigmoidf(ir + hr);
            float zz  = sigmoidf(iz + hz);
            float nn  = tanhf(inn + rr * hn);
            hv_out[(size_t)(nbase + f) * 64 + q] = (1.0f - zz) * nn + zz * h;
        }
    }
}

// ---------------------------------------------------------------------------
extern "C" void kernel_launch(void* const* d_in, const int* in_sizes, int n_in,
                              void* d_out, int out_size, void* d_ws, size_t ws_size,
                              hipStream_t stream) {
    const float* hv  = (const float*)d_in[0];   // [N,64]
    const float* he  = (const float*)d_in[1];   // [E,32]
    const int*   src = (const int*)d_in[2];     // [E]
    const int*   dst = (const int*)d_in[3];     // [E]
    const float* Wm  = (const float*)d_in[4];   // [2,160,128]
    const float* bm  = (const float*)d_in[5];   // [2,128]
    const float* Wih = (const float*)d_in[6];   // [2,192,128]
    const float* Whh = (const float*)d_in[7];   // [2,192,64]
    const float* bih = (const float*)d_in[8];   // [2,192]
    const float* bhh = (const float*)d_in[9];   // [2,192]
    float* out = (float*)d_out;                 // [N,64]

    const int N = in_sizes[0] / 64;
    const int E = in_sizes[2];
    const int n_etiles = E / 16;                // 100000
    const int n_ntiles = N / 16;                // 6250

    float* a = (float*)d_ws;                    // [N,128] mailbox

    const int    edge_lds = 16 * 160 * 2 + 32 * 4;
    const size_t gru_lds  = (size_t)(192 * 128 + 192 * 64 + 16 * 128 + 16 * 64) * 2
                          + (size_t)(2 * 16 * 192) * 4;

    for (int t = 0; t < 2; ++t) {
        const float* hv_cur = (t == 0) ? hv : out;

        int n4 = (N * 128) / 4;
        zero_kernel<<<(n4 + 255) / 256, 256, 0, stream>>>((float4*)a, n4);

        edge_kernel<<<2048, 256, edge_lds, stream>>>(
            hv_cur, he, src, dst,
            Wm + (size_t)t * 160 * 128, bm + (size_t)t * 128,
            a, n_etiles);

        gru_kernel<<<1024, 256, gru_lds, stream>>>(
            a, hv_cur, out,
            Wih + (size_t)t * 192 * 128, Whh + (size_t)t * 192 * 64,
            bih + (size_t)t * 192, bhh + (size_t)t * 192,
            n_ntiles);
    }
}
